// IFNeuron_29626684408100
// MI455X (gfx1250) — matrix operand order repaired and checked by reading, not compiled, not run on hardware
//
#include <hip/hip_runtime.h>

// ---------------------------------------------------------------------------
// IF-neuron SNN layer for MI455X (gfx1250, wave32):
//   x[t,b,o] = sum_i cur_inp[t,b,i] * weight[o,i] + bias[o]     (WMMA f16 GEMM)
//   scan over t: mem = mem*(1-spike) + x; spike = mem > 1; out = sum(spike)/T
//
// Pipeline: (1) weight f32->f16 into ws, (2) tiled WMMA GEMM (wave tile 32x32,
// 4 wmma per K-step, async global->LDS staging for W, two-phase pipelined
// staging for A), (3) register-resident IF scan.
// Workspace: X f16 [4096*2048] (16.8 MB) + W16 [2048*2048] f16 (8.4 MB).
// ---------------------------------------------------------------------------

typedef __attribute__((ext_vector_type(16))) _Float16 v16h;
typedef __attribute__((ext_vector_type(8)))  float    v8f;
typedef int v4i_vs __attribute__((vector_size(16)));   // matches builtin param type

#define T_STEPS 32
#define BATCH   128
#define DIN     2048
#define DOUT    2048
#define M_TOTAL (T_STEPS * BATCH)   // 4096 rows in the flattened GEMM

#define TILE_M 128
#define TILE_N 64
#define TILE_K 64
#define LDSP   (TILE_K + 8)   // +16B pad -> 144B row stride, conflict-free frag loads

#define AS1 __attribute__((address_space(1)))
#define AS3 __attribute__((address_space(3)))

#if defined(__has_builtin)
#if __has_builtin(__builtin_amdgcn_global_load_async_to_lds_b128)
#define HAVE_ASYNC_LDS 1
#endif
#endif

union Frag16 { int4 q[2]; v16h v; };

// ---- kernel 1: weight f32 -> f16 ------------------------------------------
__global__ __launch_bounds__(256) void convert_w_f16(const float* __restrict__ w,
                                                     _Float16* __restrict__ w16,
                                                     int n) {
  int gid = blockIdx.x * 256 + threadIdx.x;
  int stride = gridDim.x * 256;
  for (; gid < n; gid += stride) w16[gid] = (_Float16)w[gid];
}

// ---- kernel 2: x = A * W^T + bias, f16 operands, f32 accumulate -----------
__global__ __launch_bounds__(256) void snn_gemm(const float*    __restrict__ A,    // [M_TOTAL][DIN] f32
                                                const _Float16* __restrict__ W16,  // [DOUT][DIN]    f16
                                                const float*    __restrict__ bias, // [DOUT]
                                                _Float16*       __restrict__ X)    // [M_TOTAL][DOUT] f16
{
  __shared__ _Float16 As[TILE_M][LDSP];
  __shared__ _Float16 Ws[TILE_N][LDSP];

  const int tid  = threadIdx.x;
  const int lane = tid & 31;
  const int wave = tid >> 5;
  const int wm   = wave >> 1;   // 0..3 : 32-row slab
  const int wn   = wave & 1;    // 0..1 : 32-col slab
  const int lm   = lane & 15;   // index within 16
  const int lh   = lane >> 4;   // K-half selector per ISA 7.12.2

  const int m0 = blockIdx.y * TILE_M;
  const int n0 = blockIdx.x * TILE_N;

  v8f acc00 = {}, acc01 = {}, acc10 = {}, acc11 = {};

  for (int kc = 0; kc < DIN; kc += TILE_K) {
    // ---- Stage W tile first: 64x64 f16 (512 x 16B, 2/thread) via the gfx1250
    //      async global->LDS path (ASYNCcnt); runs while A loads are in flight.
#pragma unroll
    for (int i = 0; i < 2; ++i) {
      int idx = tid + i * 256;
      int row = idx >> 3;          // 8 x 16B chunks per 64-f16 row
      int c8  = (idx & 7) << 3;
#if defined(HAVE_ASYNC_LDS)
      __builtin_amdgcn_global_load_async_to_lds_b128(
          (AS1 v4i_vs*)(W16 + (size_t)(n0 + row) * DIN + kc + c8),
          (AS3 v4i_vs*)&Ws[row][c8], 0, 0);
#else
      *reinterpret_cast<int4*>(&Ws[row][c8]) =
          *reinterpret_cast<const int4*>(W16 + (size_t)(n0 + row) * DIN + kc + c8);
#endif
    }

    // ---- Stage A tile: 128x64 f32 -> f16 (2048 float4s, 8/thread, coalesced).
    // Two-phase: issue ALL global loads first (MLP), then convert + ds_store
    // in issue order so the compiler emits decreasing s_wait_loadcnt values.
    float4 av[8];
#pragma unroll
    for (int i = 0; i < 8; ++i) {
      int idx = tid + i * 256;
      int row = idx >> 4;          // 16 float4 per 64-col row
      int c4  = (idx & 15) << 2;
      av[i] = *reinterpret_cast<const float4*>(A + (size_t)(m0 + row) * DIN + kc + c4);
    }
#pragma unroll
    for (int i = 0; i < 8; ++i) {
      int idx = tid + i * 256;
      int row = idx >> 4;
      int c4  = (idx & 15) << 2;
      As[row][c4 + 0] = (_Float16)av[i].x;
      As[row][c4 + 1] = (_Float16)av[i].y;
      As[row][c4 + 2] = (_Float16)av[i].z;
      As[row][c4 + 3] = (_Float16)av[i].w;
    }

#if defined(HAVE_ASYNC_LDS)
#if __has_builtin(__builtin_amdgcn_s_wait_asynccnt)
    __builtin_amdgcn_s_wait_asynccnt(0);
#else
    asm volatile("s_wait_asynccnt 0x0" ::: "memory");
#endif
#endif
    // Prefetch next A chunk toward cache while we compute this one.
    if (kc + TILE_K < DIN) {
      __builtin_prefetch(A + (size_t)(m0 + (tid >> 1)) * DIN + (kc + TILE_K) + (tid & 1) * 32, 0, 1);
    }
    __syncthreads();

#pragma unroll
    for (int kb = 0; kb < TILE_K; kb += 32) {
      // A fragments (16x32): lane m = lm; j<8 -> k = lh*8+j, j>=8 -> k = 16+lh*8+(j-8).
      Frag16 a0, a1;
      const _Float16* ap0 = &As[wm * 32 + lm][0];
      a0.q[0] = *reinterpret_cast<const int4*>(ap0 + kb + lh * 8);
      a0.q[1] = *reinterpret_cast<const int4*>(ap0 + kb + 16 + lh * 8);
      const _Float16* ap1 = &As[wm * 32 + 16 + lm][0];
      a1.q[0] = *reinterpret_cast<const int4*>(ap1 + kb + lh * 8);
      a1.q[1] = *reinterpret_cast<const int4*>(ap1 + kb + 16 + lh * 8);

      // B fragments (32x16): lane n = lm; j<8 -> k = lh*8+j, j>=8 -> k = 16+lh*8+(j-8).
      Frag16 b0, b1;
      const _Float16* bp0 = &Ws[wn * 32 + lm][kb + lh * 8];
      b0.q[0] = *reinterpret_cast<const int4*>(bp0);
      b0.q[1] = *reinterpret_cast<const int4*>(bp0 + 16);
      const _Float16* bp1 = &Ws[wn * 32 + 16 + lm][kb + lh * 8];
      b1.q[0] = *reinterpret_cast<const int4*>(bp1);
      b1.q[1] = *reinterpret_cast<const int4*>(bp1 + 16);

      acc00 = __builtin_amdgcn_wmma_f32_16x16x32_f16(false, a0.v, false, b0.v,
                                                     (short)0, acc00, false, false);
      acc01 = __builtin_amdgcn_wmma_f32_16x16x32_f16(false, a0.v, false, b1.v,
                                                     (short)0, acc01, false, false);
      acc10 = __builtin_amdgcn_wmma_f32_16x16x32_f16(false, a1.v, false, b0.v,
                                                     (short)0, acc10, false, false);
      acc11 = __builtin_amdgcn_wmma_f32_16x16x32_f16(false, a1.v, false, b1.v,
                                                     (short)0, acc11, false, false);
    }
    __syncthreads();
  }

  // Epilogue: add bias, store f16. C/D layout: lane (lm,lh), VGPR r -> m = r + lh*8.
  const int nA = n0 + wn * 32 + lm;
  const int nB = nA + 16;
  const float bvA = bias[nA];
  const float bvB = bias[nB];
#pragma unroll
  for (int r = 0; r < 8; ++r) {
    const int m = m0 + wm * 32 + (r + lh * 8);
    X[(size_t)m * DOUT + nA]        = (_Float16)(acc00[r] + bvA);
    X[(size_t)m * DOUT + nB]        = (_Float16)(acc01[r] + bvB);
    X[(size_t)(m + 16) * DOUT + nA] = (_Float16)(acc10[r] + bvA);
    X[(size_t)(m + 16) * DOUT + nB] = (_Float16)(acc11[r] + bvB);
  }
}

// ---- kernel 3: integrate-and-fire scan over T, spike-rate output ----------
__global__ __launch_bounds__(256) void snn_scan(const _Float16* __restrict__ X,        // [T][B*DOUT] f16
                                                const float*    __restrict__ mem_init, // [B*DOUT]
                                                float*          __restrict__ out)      // [B*DOUT]
{
  const int gid = blockIdx.x * 256 + threadIdx.x;   // b*DOUT + o
  float mem   = mem_init[gid];
  float spike = 0.0f;
  float s     = 0.0f;
#pragma unroll
  for (int t = 0; t < T_STEPS; ++t) {
    float x = (float)X[(size_t)t * (BATCH * DOUT) + gid];
    mem   = mem * (1.0f - spike) + x;   // reset-by-mask + integrate
    spike = (mem > 1.0f) ? 1.0f : 0.0f; // strict > threshold
    s    += spike;
  }
  out[gid] = s * (1.0f / T_STEPS);      // * VTH / T, VTH = 1
}

// ---------------------------------------------------------------------------
extern "C" void kernel_launch(void* const* d_in, const int* in_sizes, int n_in,
                              void* d_out, int out_size, void* d_ws, size_t ws_size,
                              hipStream_t stream) {
  const float* cur_inp  = (const float*)d_in[0];  // [32,128,2048]
  const float* weight   = (const float*)d_in[1];  // [2048,2048]
  const float* bias     = (const float*)d_in[2];  // [2048]
  const float* mem_init = (const float*)d_in[3];  // [128,2048]

  _Float16* X   = (_Float16*)d_ws;                                        // 16.8 MB
  _Float16* W16 = (_Float16*)((char*)d_ws + (size_t)M_TOTAL * DOUT * 2);  // +8.4 MB

  convert_w_f16<<<1024, 256, 0, stream>>>(weight, W16, DOUT * DIN);

  dim3 grid(DOUT / TILE_N, M_TOTAL / TILE_M);  // (32, 32) blocks
  snn_gemm<<<grid, 256, 0, stream>>>(cur_inp, W16, bias, X);

  snn_scan<<<(BATCH * DOUT) / 256, 256, 0, stream>>>(X, mem_init, (float*)d_out);
}
